// OptimalTransportLoss_50362786512917
// MI455X (gfx1250) — matrix operand order repaired and checked by reading, not compiled, not run on hardware
//
#include <hip/hip_runtime.h>
#include <hip/hip_bf16.h>

// ---------------------------------------------------------------------------
// OT loss (Sinkhorn) for MI455X / gfx1250.
//  dist: 4096 x 8220 f32;  d = dist[:, 28:]  (4096 x 8192)
//  loss = _ot_loss(d) + _ot_loss(d.T)   -> scalar f32 in d_out[0]
//
// Strategy: build K1 (4096x8192) and K2 (8192x4096) in bf16 so both stay
// L2-resident (134 MB < 192 MB L2) across 50 Sinkhorn iterations.
// u-step (K v) uses v_wmma_f32_16x16x32_bf16 with a column-broadcast B.
// v-step (K^T u) is a coalesced column reduction (bandwidth optimal).
// ---------------------------------------------------------------------------

#define NN1   4096            // rows of d
#define MM1   8192            // cols of d
#define CAPC  28
#define LDD   8220            // dist row stride
#define NITER 50

typedef __attribute__((ext_vector_type(16))) __bf16 v16bf;
typedef __attribute__((ext_vector_type(8)))  __bf16 v8bf;
typedef __attribute__((ext_vector_type(8)))  float  v8f;
typedef __attribute__((ext_vector_type(8)))  unsigned short v8us;

__device__ __forceinline__ float bf2f(unsigned short h) {
    unsigned int u = ((unsigned int)h) << 16;
    float f; __builtin_memcpy(&f, &u, 4); return f;
}
__device__ __forceinline__ unsigned short f2bf(float f) {
    unsigned int u; __builtin_memcpy(&u, &f, 4);
    u += 0x7FFFu + ((u >> 16) & 1u);              // round-to-nearest-even
    return (unsigned short)(u >> 16);
}

// ---- prep: per-row maxima of the diagonal pattern --------------------------
__global__ __launch_bounds__(256) void k_d1(const float* __restrict__ dist,
                                            float* __restrict__ d1a,
                                            float* __restrict__ d1b,
                                            int* __restrict__ m1) {
    int t = blockIdx.x * 256 + threadIdx.x;
    if (t < MM1) d1b[t] = dist[(size_t)(t & (NN1 - 1)) * LDD + CAPC + t];
    if (t < NN1) {
        float x0 = dist[(size_t)t * LDD + CAPC + t];
        float x1 = dist[(size_t)t * LDD + CAPC + t + NN1];
        d1a[t] = fmaxf(x0, x1);
        m1[t]  = (x0 >= x1) ? t : t + NN1;        // argmax, ties -> first
    }
}

// ---- prep: build K1 (row-major), K2 & G2 (transposed via LDS tiles) --------
__global__ __launch_bounds__(256) void k_build(const float* __restrict__ dist,
                                               const float* __restrict__ d1a,
                                               const float* __restrict__ d1b,
                                               unsigned short* __restrict__ K1,
                                               unsigned short* __restrict__ K2,
                                               unsigned short* __restrict__ G2) {
    __shared__ float tk[32][33];
    __shared__ float tg[32][33];
    int j0 = blockIdx.x * 32;                     // column of d (row of K2)
    int i0 = blockIdx.y * 32;                     // row of d
    int tx = threadIdx.x, ty = threadIdx.y;
    float d1col = d1b[j0 + tx];
#pragma unroll
    for (int k = 0; k < 4; ++k) {
        int il = ty + 8 * k;
        int i = i0 + il, j = j0 + tx;
        float x = dist[(size_t)i * LDD + CAPC + j];
        float d1row = d1a[i];
        bool diag = (j == i) || (j == i + NN1);
        // matrix 1: GM = diag ? 0 : hinge
        float g1 = diag ? 0.f : fmaxf(0.05f + x - d1row, 0.f);
        K1[(size_t)i * MM1 + j] = f2bf(__expf(5.f * __expf(-g1)));
        // matrix 2 (on d.T): same element, d1 indexed by transpose row j
        float g2raw = diag ? fmaxf(-0.5f - x + d1col, 0.f)
                           : fmaxf(0.05f + x - d1col, 0.f);
        float g2 = diag ? 0.f : g2raw;
        tk[il][tx] = __expf(5.f * __expf(-g2));
        tg[il][tx] = g2raw;
    }
    __syncthreads();
#pragma unroll
    for (int k = 0; k < 4; ++k) {
        int jl = ty + 8 * k;
        size_t o = (size_t)(j0 + jl) * NN1 + i0 + tx;
        K2[o] = f2bf(tk[tx][jl]);
        G2[o] = f2bf(tg[tx][jl]);
    }
}

// ---- init u = 1/n ----------------------------------------------------------
__global__ __launch_bounds__(256) void k_init(float* u1, float* u2,
                                              unsigned short* u1b, unsigned short* u2b) {
    int t = blockIdx.x * 256 + threadIdx.x;
    if (t < NN1) { u1[t] = 1.f / NN1; u1b[t] = f2bf(1.f / NN1); }
    if (t < MM1) { u2[t] = 1.f / MM1; u2b[t] = f2bf(1.f / MM1); }
}

// ---- v-step main: partial K^T u (coalesced column reduction) ---------------
__global__ __launch_bounds__(256) void k_vmain(const unsigned short* __restrict__ K1,
                                               const unsigned short* __restrict__ K2,
                                               const float* __restrict__ u1,
                                               const float* __restrict__ u2,
                                               float* __restrict__ pv1,
                                               float* __restrict__ pv2) {
    const unsigned short* K; const float* u; float* part; int C, nCB, nCh;
    if (blockIdx.z == 0) { K = K1; u = u1; part = pv1; C = MM1; nCB = 4; nCh = 32; }
    else                 { K = K2; u = u2; part = pv2; C = NN1; nCB = 2; nCh = 64; }
    if ((int)blockIdx.x >= nCB || (int)blockIdx.y >= nCh) return;
    int c0 = blockIdx.x * 2048 + threadIdx.x * 8;
    int r0 = blockIdx.y * 128;
    float acc[8] = {0, 0, 0, 0, 0, 0, 0, 0};
    const unsigned short* p = K + (size_t)r0 * C + c0;
#pragma unroll 4
    for (int i = 0; i < 128; ++i) {
        float ui = u[r0 + i];                     // uniform -> scalar load
        v8us kv = *(const v8us*)p;                // 16B coalesced bf16 load
#pragma unroll
        for (int g = 0; g < 8; ++g) acc[g] = fmaf(bf2f(kv[g]), ui, acc[g]);
        p += C;
    }
    float* o = part + (size_t)blockIdx.y * C + c0;
#pragma unroll
    for (int g = 0; g < 8; ++g) o[g] = acc[g];
}

// ---- v-step epilogue: v = (1/m) / t ; also bf16 copy -----------------------
__global__ __launch_bounds__(256) void k_vepi(const float* __restrict__ pv1,
                                              const float* __restrict__ pv2,
                                              float* v1, float* v2,
                                              unsigned short* v1b, unsigned short* v2b) {
    int t = blockIdx.x * 256 + threadIdx.x;
    if (t < MM1) {
        float s = 0.f;
        for (int c = 0; c < 32; ++c) s += pv1[(size_t)c * MM1 + t];
        float val = (1.f / MM1) / s;
        v1[t] = val; v1b[t] = f2bf(val);
    } else if (t < MM1 + NN1) {
        int col = t - MM1;
        float s = 0.f;
        for (int c = 0; c < 64; ++c) s += pv2[(size_t)c * NN1 + col];
        float val = (1.f / NN1) / s;
        v2[col] = val; v2b[col] = f2bf(val);
    }
}

// ---- u-step main: partial K v via WMMA (broadcast-B matvec) ----------------
__global__ __launch_bounds__(32) void k_umain(const unsigned short* __restrict__ K1,
                                              const unsigned short* __restrict__ K2,
                                              const unsigned short* __restrict__ v1b,
                                              const unsigned short* __restrict__ v2b,
                                              float* __restrict__ pu1,
                                              float* __restrict__ pu2) {
    const unsigned short* K; const unsigned short* vb; float* part; int R, C, nT, nCh;
    if (blockIdx.z == 0) { K = K1; vb = v1b; part = pu1; R = NN1; C = MM1; nT = 256; nCh = 4; }
    else                 { K = K2; vb = v2b; part = pu2; R = MM1; C = NN1; nT = 512; nCh = 2; }
    if ((int)blockIdx.x >= nT || (int)blockIdx.y >= nCh) return;   // uniform; EXEC stays full
    int lane = threadIdx.x;
    int half = lane >> 4;                         // 0: K 0..15, 1: K 16..31
    int row  = blockIdx.x * 16 + (lane & 15);     // A-matrix row = lane&15
    int cbase = blockIdx.y * 2048;
    const unsigned short* ap = K + (size_t)row * C + cbase + half * 16;
    const unsigned short* bp = vb + cbase + half * 16;
    v8f c = {0.f, 0.f, 0.f, 0.f, 0.f, 0.f, 0.f, 0.f};
    for (int j = 0; j < 2048; j += 32) {
        v8bf alo = *(const v8bf*)(ap + j);        // A: contiguous k = 16*half + s (B matches)
        v8bf ahi = *(const v8bf*)(ap + j + 8);
        v16bf a = __builtin_shufflevector(alo, ahi,
                    0, 1, 2, 3, 4, 5, 6, 7, 8, 9, 10, 11, 12, 13, 14, 15);
        v16bf b = *(const v16bf*)(bp + j);        // B[k][q]=v[k] for all q (broadcast)
        c = __builtin_amdgcn_wmma_f32_16x16x32_bf16(false, a, false, b,
                                                    (short)0, c, false, false);
    }
    // column N=0 of D: lane 0 -> M=0..7, lane 16 -> M=8..15
    if ((lane & 15) == 0) {
        float* o = part + (size_t)blockIdx.y * R + blockIdx.x * 16 + half * 8;
#pragma unroll
        for (int g = 0; g < 8; ++g) o[g] = c[g];
    }
}

// ---- u-step epilogue: u = 1/(n * Kv) ; also bf16 copy ----------------------
__global__ __launch_bounds__(256) void k_uepi(const float* __restrict__ pu1,
                                              const float* __restrict__ pu2,
                                              float* u1, float* u2,
                                              unsigned short* u1b, unsigned short* u2b) {
    int t = blockIdx.x * 256 + threadIdx.x;
    if (t < NN1) {
        float s = 0.f;
        for (int c = 0; c < 4; ++c) s += pu1[(size_t)c * NN1 + t];
        float val = 1.f / ((float)NN1 * s);
        u1[t] = val; u1b[t] = f2bf(val);
    } else if (t < NN1 + MM1) {
        int row = t - NN1;
        float s = 0.f;
        for (int c = 0; c < 2; ++c) s += pu2[(size_t)c * MM1 + row];
        float val = 1.f / ((float)MM1 * s);
        u2[row] = val; u2b[row] = f2bf(val);
    }
}

// ---- final per-row loss, matrix 1 (maxel + reweight collapse to one norm) --
__global__ __launch_bounds__(256) void k_f1(const float* __restrict__ dist,
                                            const unsigned short* __restrict__ K1,
                                            const float* __restrict__ v1,
                                            const float* __restrict__ d1a,
                                            const int* __restrict__ m1,
                                            float* __restrict__ q) {
    int i = blockIdx.x;
    float d1 = d1a[i]; int mi = m1[i];
    float A = 0.f, B = 0.f;
    for (int j = threadIdx.x; j < MM1; j += 256) {
        bool kept = ((j < NN1) || (j == i + NN1)) && (j != mi);
        if (kept) {
            float kv = bf2f(K1[(size_t)i * MM1 + j]) * v1[j];
            float x  = dist[(size_t)i * LDD + CAPC + j];
            bool diag = (j == i) || (j == i + NN1);
            float gm = diag ? fmaxf(-0.5f - x + d1, 0.f)
                            : fmaxf(0.05f + x - d1, 0.f);
            A = fmaf(gm, kv, A); B += kv;
        }
    }
    __shared__ float sA[256], sB[256];
    sA[threadIdx.x] = A; sB[threadIdx.x] = B; __syncthreads();
    for (int s = 128; s > 0; s >>= 1) {
        if (threadIdx.x < s) { sA[threadIdx.x] += sA[threadIdx.x + s];
                               sB[threadIdx.x] += sB[threadIdx.x + s]; }
        __syncthreads();
    }
    if (threadIdx.x == 0) q[i] = sA[0] / sB[0];
}

// ---- final per-row loss, matrix 2 (sole diag == maxel -> always excluded) --
__global__ __launch_bounds__(256) void k_f2(const unsigned short* __restrict__ K2,
                                            const unsigned short* __restrict__ G2,
                                            const float* __restrict__ v2,
                                            float* __restrict__ q) {
    int r = blockIdx.x;
    int skip = r & (NN1 - 1);
    float A = 0.f, B = 0.f;
    for (int j = threadIdx.x; j < NN1; j += 256) {
        if (j == skip) continue;
        size_t o = (size_t)r * NN1 + j;
        float kv = bf2f(K2[o]) * v2[j];
        A = fmaf(bf2f(G2[o]), kv, A); B += kv;
    }
    __shared__ float sA[256], sB[256];
    sA[threadIdx.x] = A; sB[threadIdx.x] = B; __syncthreads();
    for (int s = 128; s > 0; s >>= 1) {
        if (threadIdx.x < s) { sA[threadIdx.x] += sA[threadIdx.x + s];
                               sB[threadIdx.x] += sB[threadIdx.x + s]; }
        __syncthreads();
    }
    if (threadIdx.x == 0) q[NN1 + r] = sA[0] / sB[0];
}

// ---- deterministic final reduction -----------------------------------------
__global__ __launch_bounds__(256) void k_f3(const float* __restrict__ q, float* out) {
    __shared__ float s[256];
    float a = 0.f;
    for (int t = threadIdx.x; t < NN1 + MM1; t += 256) a += q[t];
    s[threadIdx.x] = a; __syncthreads();
    for (int w = 128; w > 0; w >>= 1) {
        if (threadIdx.x < w) s[threadIdx.x] += s[threadIdx.x + w];
        __syncthreads();
    }
    if (threadIdx.x == 0) out[0] = s[0];
}

extern "C" void kernel_launch(void* const* d_in, const int* in_sizes, int n_in,
                              void* d_out, int out_size, void* d_ws, size_t ws_size,
                              hipStream_t stream) {
    (void)in_sizes; (void)n_in; (void)out_size; (void)ws_size;
    const float* dist = (const float*)d_in[0];    // labels (d_in[1]) unused
    char* ws = (char*)d_ws;

    unsigned short* K1 = (unsigned short*)(ws);                       // 64 MiB
    unsigned short* K2 = (unsigned short*)(ws + (64ull << 20));       // 64 MiB
    unsigned short* G2 = (unsigned short*)(ws + (128ull << 20));      // 64 MiB
    const size_t V = 192ull << 20;
    float* d1a = (float*)(ws + V);
    float* d1b = (float*)(ws + V + 16384);
    int*   m1  = (int*)  (ws + V + 49152);
    float* u1  = (float*)(ws + V + 65536);
    float* v1  = (float*)(ws + V + 81920);
    float* u2  = (float*)(ws + V + 114688);
    float* v2  = (float*)(ws + V + 147456);
    unsigned short* u1b = (unsigned short*)(ws + V + 163840);
    unsigned short* u2b = (unsigned short*)(ws + V + 172032);
    unsigned short* v1b = (unsigned short*)(ws + V + 188416);
    unsigned short* v2b = (unsigned short*)(ws + V + 204800);
    float* pv1 = (float*)(ws + V + 212992);
    float* pv2 = (float*)(ws + V + 1261568);
    float* pu1 = (float*)(ws + V + 2310144);
    float* pu2 = (float*)(ws + V + 2375680);
    float* q   = (float*)(ws + V + 2441216);

    k_d1   <<<32, 256, 0, stream>>>(dist, d1a, d1b, m1);
    k_build<<<dim3(256, 128), dim3(32, 8), 0, stream>>>(dist, d1a, d1b, K1, K2, G2);
    k_init <<<32, 256, 0, stream>>>(u1, u2, u1b, u2b);

    for (int it = 0; it < NITER; ++it) {
        k_vmain<<<dim3(4, 64, 2), 256, 0, stream>>>(K1, K2, u1, u2, pv1, pv2);
        k_vepi <<<48, 256, 0, stream>>>(pv1, pv2, v1, v2, v1b, v2b);
        if (it != NITER - 1) {                    // last u cancels in the row norm
            k_umain<<<dim3(512, 4, 2), 32, 0, stream>>>(K1, K2, v1b, v2b, pu1, pu2);
            k_uepi <<<48, 256, 0, stream>>>(pu1, pu2, u1, u2, u1b, u2b);
        }
    }

    k_f1<<<NN1, 256, 0, stream>>>(dist, K1, v1, d1a, m1, q);
    k_f2<<<MM1, 256, 0, stream>>>(K2, G2, v2, q);
    k_f3<<<1, 256, 0, stream>>>(q, (float*)d_out);
}
